// PureMambaBlock_56221121904989
// MI455X (gfx1250) — matrix operand order repaired and checked by reading, not compiled, not run on hardware
//
#include <hip/hip_runtime.h>
#include <hip/hip_bf16.h>
#include <math.h>

// ---------------------------------------------------------------------------
// Mamba block forward for MI455X (gfx1250, wave32, WMMA).
// GEMMs: bf16 operands (converted once), fp32 accumulate via
// v_wmma_f32_16x16x32_bf16. Block tile 64x128, wave tile 32x32, K-step 64
// -> 8 WMMAs per barrier pair. LDS staged with GLOBAL_LOAD_ASYNC_TO_LDS_B128
// (ASYNCcnt) so no VGPR round-trip on the fill path.
// ---------------------------------------------------------------------------

typedef __attribute__((ext_vector_type(16))) __bf16         v16bf;
typedef __attribute__((ext_vector_type(8)))  float          v8f;
typedef __attribute__((ext_vector_type(8)))  unsigned short u16x8;
typedef __attribute__((ext_vector_type(16))) unsigned short u16x16;

#define D_MODEL 1024
#define D_STATE 16
#define D_CONV  4
#define D_INNER 2048
#define DT_RANK 64
#define SEQ     2048
#define XDBL_N  (DT_RANK + 2 * D_STATE)   // 96
#define NCHUNK  32
#define CHUNK   (SEQ / NCHUNK)            // 64
#define LDS_STRIDE 72                     // 64 halfs + 8 pad, 144B (16B-aligned)

__device__ __forceinline__ unsigned short f2bf(float f) {
    // round-to-nearest-even fp32 -> bf16
    unsigned int u = __float_as_uint(f);
    u += 0x7FFFu + ((u >> 16) & 1u);
    return (unsigned short)(u >> 16);
}

// Direct global -> LDS async copy, 16B per lane (CDNA5 VGLOBAL opcode 98).
// vdst = per-lane LDS byte offset, vaddr = per-lane 64-bit global address.
__device__ __forceinline__ void async_ld_b128(unsigned lds_off, const void* g) {
    asm volatile("global_load_async_to_lds_b128 %0, %1, off"
                 :: "v"(lds_off), "v"(g) : "memory");
}
__device__ __forceinline__ void wait_async0() {
    asm volatile("s_wait_asynccnt 0x0" ::: "memory");
}
__device__ __forceinline__ unsigned lds_off_of(const void* p) {
    // addrspace(3) objects map to flat {SHARED_BASE, lds_offset}: low 32 bits
    // are the raw LDS byte offset the async-load VDST operand wants.
    return (unsigned)(size_t)p;
}

// ---------------------------------------------------------------------------
// Bulk fp32 -> bf16 conversion (streaming, memory bound). n % 8 == 0.
// ---------------------------------------------------------------------------
__global__ void cvt_f32_bf16_kernel(const float* __restrict__ src,
                                    unsigned short* __restrict__ dst, int n) {
    const int i = (blockIdx.x * blockDim.x + threadIdx.x) * 8;
    if (i >= n) return;
    const float4 a = *reinterpret_cast<const float4*>(src + i);
    const float4 b = *reinterpret_cast<const float4*>(src + i + 4);
    u16x8 o;
    o[0] = f2bf(a.x); o[1] = f2bf(a.y); o[2] = f2bf(a.z); o[3] = f2bf(a.w);
    o[4] = f2bf(b.x); o[5] = f2bf(b.y); o[6] = f2bf(b.z); o[7] = f2bf(b.w);
    *reinterpret_cast<u16x8*>(dst + i) = o;
}

// ---------------------------------------------------------------------------
// GEMM: C[M,N] = A[M,K] * W[N,K]^T, A/W pre-converted bf16, C fp32.
// Block: 256 threads = 8 waves (2 M x 4 N), block tile 64x128, K-step 64.
// Each wave: 32x32 output = 4 accumulators, 8 WMMAs per K round.
// EP==1: fused bias + softplus epilogue. ASYNC==1: B tile staged with
// async-to-LDS (requires N % 128 == 0); A tile always async (M % 64 == 0).
// ---------------------------------------------------------------------------
template <int EP, int ASYNC>
__global__ void __launch_bounds__(256)
gemm_wmma_bf16(const unsigned short* __restrict__ A, int sA,
               const unsigned short* __restrict__ W,
               float* __restrict__ C, int sC,
               int N, int K,
               const float* __restrict__ bias) {
    __shared__ alignas(16) unsigned short As[64][LDS_STRIDE];
    __shared__ alignas(16) unsigned short Bs[128][LDS_STRIDE];

    const int tid  = threadIdx.x;
    const int lane = tid & 31;
    const int w    = tid >> 5;
    const int wm   = w & 1;          // 2 M sub-blocks of 32
    const int wn   = w >> 1;         // 4 N sub-blocks of 32
    const int m0   = blockIdx.y * 64;
    const int n0   = blockIdx.x * 128;

    v8f acc[2][2];
#pragma unroll
    for (int mi = 0; mi < 2; ++mi)
#pragma unroll
        for (int ni = 0; ni < 2; ++ni)
            acc[mi][ni] = (v8f){0.f,0.f,0.f,0.f,0.f,0.f,0.f,0.f};

    const int h   = lane >> 4;       // K-half select within a fragment
    const int l15 = lane & 15;

    for (int k0 = 0; k0 < K; k0 += 64) {
        // ---- prefetch next K tiles into the near cache ----
        if (k0 + 64 < K) {
            __builtin_prefetch(A + (size_t)(m0 + (tid >> 2)) * sA + k0 + 64, 0, 3);
            __builtin_prefetch(W + (size_t)(n0 + (tid >> 1)) * K  + k0 + 64, 0, 3);
        }
        // ---- stage A tile: 64 rows x 64 halfs, 2 async 16B chunks/thread ----
#pragma unroll
        for (int i = 0; i < 2; ++i) {
            const int c   = tid * 2 + i;
            const int r   = c >> 3;            // 0..63
            const int off = (c & 7) * 8;       // 0..56
            async_ld_b128(lds_off_of(&As[r][off]),
                          A + (size_t)(m0 + r) * sA + k0 + off);
        }
        // ---- stage B tile: 128 rows x 64 halfs, 4 chunks/thread ----
#pragma unroll
        for (int i = 0; i < 4; ++i) {
            const int c   = tid * 4 + i;
            const int r   = c >> 3;            // 0..127
            const int off = (c & 7) * 8;
            if (ASYNC) {
                async_ld_b128(lds_off_of(&Bs[r][off]),
                              W + (size_t)(n0 + r) * K + k0 + off);
            } else {
                u16x8 v = {0, 0, 0, 0, 0, 0, 0, 0};
                if (n0 + r < N)
                    v = *reinterpret_cast<const u16x8*>(
                        W + (size_t)(n0 + r) * K + k0 + off);
                *reinterpret_cast<u16x8*>(&Bs[r][off]) = v;
            }
        }
        wait_async0();                // our async fills done; barrier covers rest
        __syncthreads();

        // ---- 2 K-windows of 32, 8 WMMAs ----
#pragma unroll
        for (int ks = 0; ks < 2; ++ks) {
            v16bf a[2], b[2];
#pragma unroll
            for (int mi = 0; mi < 2; ++mi) {
                const unsigned short* ap = &As[wm * 32 + mi * 16 + l15][ks * 32];
                u16x8 lo = *reinterpret_cast<const u16x8*>(ap + h * 8);
                u16x8 hi = *reinterpret_cast<const u16x8*>(ap + 16 + h * 8);
                a[mi] = __builtin_bit_cast(v16bf, (u16x16)__builtin_shufflevector(
                            lo, hi, 0,1,2,3,4,5,6,7,8,9,10,11,12,13,14,15));
            }
#pragma unroll
            for (int ni = 0; ni < 2; ++ni) {
                const unsigned short* bp = &Bs[wn * 32 + ni * 16 + l15][ks * 32];
                u16x8 lo = *reinterpret_cast<const u16x8*>(bp + h * 8);
                u16x8 hi = *reinterpret_cast<const u16x8*>(bp + 16 + h * 8);
                b[ni] = __builtin_bit_cast(v16bf, (u16x16)__builtin_shufflevector(
                            lo, hi, 0,1,2,3,4,5,6,7,8,9,10,11,12,13,14,15));
            }
#pragma unroll
            for (int mi = 0; mi < 2; ++mi)
#pragma unroll
                for (int ni = 0; ni < 2; ++ni)
                    acc[mi][ni] = __builtin_amdgcn_wmma_f32_16x16x32_bf16(
                        false, a[mi], false, b[ni], (short)0, acc[mi][ni],
                        false, false);
        }
        __syncthreads();
    }

    // ---- store: C/D layout elem r -> M = h*8 + r ----
#pragma unroll
    for (int ni = 0; ni < 2; ++ni) {
        const int nn = n0 + wn * 32 + ni * 16 + l15;
        if (nn < N) {
#pragma unroll
            for (int mi = 0; mi < 2; ++mi) {
#pragma unroll
                for (int r = 0; r < 8; ++r) {
                    const int mm = m0 + wm * 32 + mi * 16 + h * 8 + r;
                    float v = acc[mi][ni][r];
                    if (EP == 1) {
                        v += bias[nn];
                        v = (v > 20.f) ? v : log1pf(expf(v));
                    }
                    C[(size_t)mm * sC + nn] = v;
                }
            }
        }
    }
}

// ---------------------------------------------------------------------------
// Depthwise causal conv1d (k=4) + bias + SiLU; emits fp32 (for scan) and
// bf16 (for the x_dbl GEMM).
// ---------------------------------------------------------------------------
__global__ void conv_silu_kernel(const float* __restrict__ xz,
                                 const float* __restrict__ cw,
                                 const float* __restrict__ cb,
                                 float* __restrict__ out_f,
                                 unsigned short* __restrict__ out_bf) {
    const int idx = blockIdx.x * blockDim.x + threadIdx.x;   // SEQ*D_INNER
    const int c = idx & (D_INNER - 1);
    const int t = idx >> 11;
    float acc = cb[c];
#pragma unroll
    for (int j = 0; j < D_CONV; ++j) {
        const int tt = t + j - (D_CONV - 1);
        if (tt >= 0)
            acc += xz[(size_t)tt * (2 * D_INNER) + c] * cw[c * D_CONV + j];
    }
    const float s = acc / (1.f + __expf(-acc));              // silu
    out_f[idx]  = s;
    out_bf[idx] = f2bf(s);
}

// ---------------------------------------------------------------------------
// Chunked selective scan, pass 1: per (chunk, channel) partials.
// P = exp(A * sum dt) (chunk-composed decay), Q = chunk state from h0=0.
// ---------------------------------------------------------------------------
__global__ void scan_partial_kernel(const float* __restrict__ xc,
                                    const float* __restrict__ dtm,
                                    const float* __restrict__ xdbl,
                                    const float* __restrict__ A_log,
                                    float* __restrict__ P,
                                    float* __restrict__ Q) {
    const int gid   = blockIdx.x * blockDim.x + threadIdx.x; // NCHUNK*D_INNER
    const int d     = gid & (D_INNER - 1);
    const int chunk = gid >> 11;

    float A[D_STATE], hh[D_STATE];
#pragma unroll
    for (int n = 0; n < D_STATE; ++n) {
        A[n]  = -__expf(A_log[d * D_STATE + n]);
        hh[n] = 0.f;
    }
    float sdt = 0.f;
    const int t0 = chunk * CHUNK;
    for (int i = 0; i < CHUNK; ++i) {
        const int   t   = t0 + i;
        const float xt  = xc [(size_t)t * D_INNER + d];
        const float dtt = dtm[(size_t)t * D_INNER + d];
        const float* bp = xdbl + (size_t)t * XDBL_N + DT_RANK;
        sdt += dtt;
#pragma unroll
        for (int n = 0; n < D_STATE; ++n) {
            const float dA = __expf(A[n] * dtt);
            hh[n] = dA * hh[n] + (dtt * bp[n]) * xt;
        }
    }
    const size_t base = ((size_t)chunk * D_INNER + d) * D_STATE;
#pragma unroll
    for (int n = 0; n < D_STATE; ++n) {
        P[base + n] = __expf(A[n] * sdt);
        Q[base + n] = hh[n];
    }
}

// pass 2: serial carry across the 32 chunks, per (d, n) recurrence.
__global__ void scan_carry_kernel(const float* __restrict__ P,
                                  const float* __restrict__ Q,
                                  float* __restrict__ H0) {
    const int gid = blockIdx.x * blockDim.x + threadIdx.x;   // D_INNER*D_STATE
    float hcar = 0.f;
    for (int c = 0; c < NCHUNK; ++c) {
        const size_t idx = (size_t)c * D_INNER * D_STATE + gid;
        H0[idx] = hcar;
        hcar = P[idx] * hcar + Q[idx];
    }
}

// pass 3: replay each chunk with its carry-in; emit bf16 y*silu(z) directly
// (sole consumer is the bf16 output GEMM).
__global__ void scan_final_kernel(const float* __restrict__ xc,
                                  const float* __restrict__ dtm,
                                  const float* __restrict__ xdbl,
                                  const float* __restrict__ A_log,
                                  const float* __restrict__ Dp,
                                  const float* __restrict__ xz,
                                  const float* __restrict__ H0,
                                  unsigned short* __restrict__ ybf) {
    const int gid   = blockIdx.x * blockDim.x + threadIdx.x; // NCHUNK*D_INNER
    const int d     = gid & (D_INNER - 1);
    const int chunk = gid >> 11;

    float A[D_STATE], hh[D_STATE];
    const size_t base = ((size_t)chunk * D_INNER + d) * D_STATE;
#pragma unroll
    for (int n = 0; n < D_STATE; ++n) {
        A[n]  = -__expf(A_log[d * D_STATE + n]);
        hh[n] = H0[base + n];
    }
    const float Dd = Dp[d];
    const int   t0 = chunk * CHUNK;
    for (int i = 0; i < CHUNK; ++i) {
        const int   t   = t0 + i;
        const float xt  = xc [(size_t)t * D_INNER + d];
        const float dtt = dtm[(size_t)t * D_INNER + d];
        const float* bp = xdbl + (size_t)t * XDBL_N + DT_RANK;
        const float* cp = bp + D_STATE;
        float yt = 0.f;
#pragma unroll
        for (int n = 0; n < D_STATE; ++n) {
            const float dA = __expf(A[n] * dtt);
            hh[n] = dA * hh[n] + (dtt * bp[n]) * xt;
            yt   += hh[n] * cp[n];
        }
        const float z = xz[(size_t)t * (2 * D_INNER) + D_INNER + d];
        yt = (yt + Dd * xt) * (z / (1.f + __expf(-z)));
        ybf[(size_t)t * D_INNER + d] = f2bf(yt);
    }
}

// ---------------------------------------------------------------------------
extern "C" void kernel_launch(void* const* d_in, const int* in_sizes, int n_in,
                              void* d_out, int out_size, void* d_ws, size_t ws_size,
                              hipStream_t stream) {
    (void)in_sizes; (void)n_in; (void)out_size; (void)ws_size;
    const float* x     = (const float*)d_in[0];
    const float* W_in  = (const float*)d_in[1];
    const float* convw = (const float*)d_in[2];
    const float* convb = (const float*)d_in[3];
    const float* W_x   = (const float*)d_in[4];
    const float* W_dt  = (const float*)d_in[5];
    const float* b_dt  = (const float*)d_in[6];
    const float* A_log = (const float*)d_in[7];
    const float* Dp    = (const float*)d_in[8];
    const float* W_out = (const float*)d_in[9];
    float* out = (float*)d_out;

    // ---- workspace carve: fp32 region first, then bf16 region ----
    float* ws    = (float*)d_ws;
    float* xz    = ws;                                   // SEQ * 2*D_INNER
    float* xc    = xz    + (size_t)SEQ * 2 * D_INNER;    // SEQ * D_INNER
    float* xdbl  = xc    + (size_t)SEQ * D_INNER;        // SEQ * 96
    float* dtbuf = xdbl  + (size_t)SEQ * XDBL_N;         // SEQ * D_INNER
    float* Pbuf  = dtbuf + (size_t)SEQ * D_INNER;        // NCHUNK*D_INNER*16
    float* Qbuf  = Pbuf  + (size_t)NCHUNK * D_INNER * D_STATE;
    float* H0buf = Qbuf  + (size_t)NCHUNK * D_INNER * D_STATE;
    unsigned short* bws  = (unsigned short*)(H0buf + (size_t)NCHUNK * D_INNER * D_STATE);
    unsigned short* xbf    = bws;                                  // 2M
    unsigned short* Winb   = xbf    + (size_t)SEQ * D_MODEL;       // 4M
    unsigned short* Wxb    = Winb   + (size_t)2 * D_INNER * D_MODEL;
    unsigned short* Wdtb   = Wxb    + (size_t)XDBL_N * D_INNER;
    unsigned short* Woutb  = Wdtb   + (size_t)D_INNER * DT_RANK;
    unsigned short* xcbf   = Woutb  + (size_t)D_MODEL * D_INNER;
    unsigned short* xdblbf = xcbf   + (size_t)SEQ * D_INNER;
    unsigned short* ybf    = xdblbf + (size_t)SEQ * XDBL_N;

    const int CT = 256;
    auto cvt = [&](const float* s, unsigned short* dst, size_t n) {
        cvt_f32_bf16_kernel<<<(unsigned)(n / 8 + CT - 1) / CT, CT, 0, stream>>>(
            s, dst, (int)n);
    };

    // ---- one-time conversions (weights + x) ----
    cvt(x,     xbf,   (size_t)SEQ * D_MODEL);
    cvt(W_in,  Winb,  (size_t)2 * D_INNER * D_MODEL);
    cvt(W_x,   Wxb,   (size_t)XDBL_N * D_INNER);
    cvt(W_dt,  Wdtb,  (size_t)D_INNER * DT_RANK);
    cvt(W_out, Woutb, (size_t)D_MODEL * D_INNER);

    // 1. xz = x @ W_in^T : (2048 x 4096), K=1024
    gemm_wmma_bf16<0, 1><<<dim3((2 * D_INNER) / 128, SEQ / 64), 256, 0, stream>>>(
        xbf, D_MODEL, Winb, xz, 2 * D_INNER, 2 * D_INNER, D_MODEL, nullptr);

    // 2. causal depthwise conv + silu (fp32 + bf16 outputs)
    conv_silu_kernel<<<(SEQ * D_INNER) / CT, CT, 0, stream>>>(
        xz, convw, convb, xc, xcbf);

    // 3. xdbl = xc @ W_x^T : (2048 x 96), K=2048 (N edge guarded, sync B path)
    gemm_wmma_bf16<0, 0><<<dim3(1, SEQ / 64), 256, 0, stream>>>(
        xcbf, D_INNER, Wxb, xdbl, XDBL_N, XDBL_N, D_INNER, nullptr);
    cvt(xdbl, xdblbf, (size_t)SEQ * XDBL_N);

    // 4. dt = softplus(xdbl[:, :64] @ W_dt^T + b_dt) : (2048 x 2048), K=64
    gemm_wmma_bf16<1, 1><<<dim3(D_INNER / 128, SEQ / 64), 256, 0, stream>>>(
        xdblbf, XDBL_N, Wdtb, dtbuf, D_INNER, D_INNER, DT_RANK, b_dt);

    // 5. chunked selective scan (3 passes) + silu(z) gating -> bf16 y
    scan_partial_kernel<<<(NCHUNK * D_INNER) / CT, CT, 0, stream>>>(
        xc, dtbuf, xdbl, A_log, Pbuf, Qbuf);
    scan_carry_kernel<<<(D_INNER * D_STATE) / CT, CT, 0, stream>>>(
        Pbuf, Qbuf, H0buf);
    scan_final_kernel<<<(NCHUNK * D_INNER) / CT, CT, 0, stream>>>(
        xc, dtbuf, xdbl, A_log, Dp, xz, H0buf, ybf);

    // 6. out = y @ W_out^T : (2048 x 1024), K=2048
    gemm_wmma_bf16<0, 1><<<dim3(D_MODEL / 128, SEQ / 64), 256, 0, stream>>>(
        ybf, D_INNER, Woutb, out, D_MODEL, D_MODEL, D_INNER, nullptr);
}
